// Head_22007412424942
// MI455X (gfx1250) — matrix operand order repaired and checked by reading, not compiled, not run on hardware
//
#include <hip/hip_runtime.h>
#include <hip/hip_bf16.h>

typedef __bf16 bf16_t;
typedef __attribute__((ext_vector_type(16))) __bf16 v16bf;
typedef __attribute__((ext_vector_type(8)))  __bf16 v8bf;
typedef __attribute__((ext_vector_type(8)))  float  v8f;

#define B_DIM 4
#define T_DIM 4096
#define C_DIM 1024
#define H_DIM 64

__device__ __forceinline__ v16bf cat8(v8bf a, v8bf b) {
  return __builtin_shufflevector(a, b, 0,1,2,3,4,5,6,7,8,9,10,11,12,13,14,15);
}

// ---------------- Kernel 1: W [C,H] fp32 -> Wt [H,C] bf16 ----------------
__global__ void wtrans_kernel(const float* __restrict__ W, bf16_t* __restrict__ Wt) {
  int idx = blockIdx.x * blockDim.x + threadIdx.x;   // over H*C
  int h = idx / C_DIM;
  int c = idx % C_DIM;
  Wt[idx] = (bf16_t)W[(size_t)c * H_DIM + h];
}

// ---------------- Kernel 2: QKV projection via WMMA ----------------
// one wave per 16-row tile of x; outputs q,k bf16 [B*T, H], v transposed bf16 [B][H][T]
__global__ void __launch_bounds__(256)
qkv_kernel(const float* __restrict__ x,
           const bf16_t* __restrict__ wqt, const bf16_t* __restrict__ wkt,
           const bf16_t* __restrict__ wvt,
           const float* __restrict__ bq, const float* __restrict__ bk,
           const float* __restrict__ bv,
           bf16_t* __restrict__ qb, bf16_t* __restrict__ kb,
           bf16_t* __restrict__ vtb)
{
  const int wave = (blockIdx.x * blockDim.x + threadIdx.x) >> 5;
  const int lane = threadIdx.x & 31;
  const int R0   = wave * 16;                 // row base in [0, B*T)
  const int mrow = lane & 15;
  const int hl   = lane >> 4;                 // half-wave id
  const int kbaseA = hl ? 8 : 0;              // A-frag K start (within 32-chunk)
  const int koffB  = hl ? 8 : 0;              // B-frag K start

  v8f acc[3][4];
#pragma unroll
  for (int m = 0; m < 3; ++m)
#pragma unroll
    for (int t = 0; t < 4; ++t)
#pragma unroll
      for (int e = 0; e < 8; ++e) acc[m][t][e] = 0.0f;

  const bf16_t* wt[3] = {wqt, wkt, wvt};

  for (int kc = 0; kc < C_DIM / 32; ++kc) {
    // A fragment: 16 fp32 of x -> bf16 (two contiguous 8-runs)
    const float* xp = x + (size_t)(R0 + mrow) * C_DIM + kc * 32 + kbaseA;
    v8f xa = *(const v8f*)(xp);
    v8f xb = *(const v8f*)(xp + 16);
    v16bf a;
#pragma unroll
    for (int e = 0; e < 8; ++e) { a[e] = (bf16_t)xa[e]; a[e + 8] = (bf16_t)xb[e]; }

#pragma unroll
    for (int m = 0; m < 3; ++m) {
#pragma unroll
      for (int t = 0; t < 4; ++t) {
        const bf16_t* wp = wt[m] + (size_t)(t * 16 + mrow) * C_DIM + kc * 32 + koffB;
        v8bf b0 = *(const v8bf*)(wp);
        v8bf b1 = *(const v8bf*)(wp + 16);
        v16bf bf = cat8(b0, b1);
        acc[m][t] = __builtin_amdgcn_wmma_f32_16x16x32_bf16(
            false, a, false, bf, (short)0, acc[m][t], false, false);
      }
    }
  }

  // epilogue: bias + stores
  const float* bias[3] = {bq, bk, bv};
  const int b   = R0 / T_DIM;
  const int tp0 = R0 % T_DIM;
#pragma unroll
  for (int m = 0; m < 3; ++m) {
#pragma unroll
    for (int t = 0; t < 4; ++t) {
      const int h = t * 16 + mrow;
      const float bb = bias[m][h];
#pragma unroll
      for (int r = 0; r < 8; ++r) {
        const float val = acc[m][t][r] + bb;
        const int row = R0 + r + 8 * hl;
        if (m == 0)      qb[(size_t)row * H_DIM + h] = (bf16_t)val;
        else if (m == 1) kb[(size_t)row * H_DIM + h] = (bf16_t)val;
        else             vtb[(size_t)b * H_DIM * T_DIM + (size_t)h * T_DIM
                             + (tp0 + r + 8 * hl)] = (bf16_t)val;
      }
    }
  }
}

// ---------------- Kernel 3: flash attention, one wave per 16 query rows ----------------
__global__ void __launch_bounds__(256)
attn_kernel(const bf16_t* __restrict__ qb, const bf16_t* __restrict__ kb,
            const bf16_t* __restrict__ vtb, float* __restrict__ out)
{
  const float scale = 0.03125f;               // C^-0.5 = 1024^-0.5
  __shared__ __align__(16) bf16_t lds_p[8][16 * 32];

  const int wid  = threadIdx.x >> 5;
  const int lane = threadIdx.x & 31;
  const int gw   = blockIdx.x * 8 + wid;
  const int b    = gw / (T_DIM / 16);
  const int qt   = gw % (T_DIM / 16);
  const int qbase = qt * 16;
  const int mrow = lane & 15;
  const int hl   = lane >> 4;
  const int kbaseA = hl ? 8 : 0;
  const int koffB  = hl ? 8 : 0;

  // Q A-fragments, 2 chunks of K=32 over H=64
  v16bf qf[2];
  {
    const bf16_t* qrow = qb + ((size_t)b * T_DIM + qbase + mrow) * H_DIM;
#pragma unroll
    for (int c = 0; c < 2; ++c) {
      v8bf a0 = *(const v8bf*)(qrow + c * 32 + kbaseA);
      v8bf a1 = *(const v8bf*)(qrow + c * 32 + kbaseA + 16);
      qf[c] = cat8(a0, a1);
    }
  }

  v8f o[4];
#pragma unroll
  for (int t = 0; t < 4; ++t)
#pragma unroll
    for (int e = 0; e < 8; ++e) o[t][e] = 0.0f;

  float mrun[8], lrun[8];
#pragma unroll
  for (int r = 0; r < 8; ++r) { mrun[r] = -__builtin_inff(); lrun[r] = 0.0f; }

  bf16_t* lp = &lds_p[wid][0];

  for (int sbase = 0; sbase < qbase + 16; sbase += 32) {
    // ---- S = Q K^T (16 x 32) ----
    v8f s[2];
#pragma unroll
    for (int n2 = 0; n2 < 2; ++n2)
#pragma unroll
      for (int e = 0; e < 8; ++e) s[n2][e] = 0.0f;

#pragma unroll
    for (int n2 = 0; n2 < 2; ++n2) {
      const bf16_t* kp = kb + ((size_t)b * T_DIM + sbase + n2 * 16 + mrow) * H_DIM;
#pragma unroll
      for (int c = 0; c < 2; ++c) {
        v8bf b0 = *(const v8bf*)(kp + c * 32 + koffB);
        v8bf b1 = *(const v8bf*)(kp + c * 32 + koffB + 16);
        v16bf kf = cat8(b0, b1);
        s[n2] = __builtin_amdgcn_wmma_f32_16x16x32_bf16(
            false, qf[c], false, kf, (short)0, s[n2], false, false);
      }
    }

    // ---- scale + causal mask (selects only; EXEC stays all-ones) ----
    const bool need_mask = (sbase + 31 > qbase);
#pragma unroll
    for (int n2 = 0; n2 < 2; ++n2) {
#pragma unroll
      for (int r = 0; r < 8; ++r) {
        float v = s[n2][r] * scale;
        if (need_mask) {
          const int sidx = sbase + n2 * 16 + mrow;
          const int ridx = qbase + r + 8 * hl;
          v = (sidx > ridx) ? -__builtin_inff() : v;
        }
        s[n2][r] = v;
      }
    }

    // ---- online softmax (row reductions across 16-lane halves) ----
    float corr[8];
    v8f p[2];
#pragma unroll
    for (int r = 0; r < 8; ++r) {
      float mx = fmaxf(s[0][r], s[1][r]);
#pragma unroll
      for (int d = 1; d < 16; d <<= 1) mx = fmaxf(mx, __shfl_xor(mx, d, 32));
      const float mnew = fmaxf(mrun[r], mx);
      const float p0 = __expf(s[0][r] - mnew);
      const float p1 = __expf(s[1][r] - mnew);
      float ps = p0 + p1;
#pragma unroll
      for (int d = 1; d < 16; d <<= 1) ps += __shfl_xor(ps, d, 32);
      corr[r] = __expf(mrun[r] - mnew);
      lrun[r] = lrun[r] * corr[r] + ps;
      mrun[r] = mnew;
      p[0][r] = p0; p[1][r] = p1;
    }

    // rescale running O
#pragma unroll
    for (int t = 0; t < 4; ++t)
#pragma unroll
      for (int r = 0; r < 8; ++r) o[t][r] *= corr[r];

    // ---- P (C/D layout) -> LDS row-major -> A-fragment ----
#pragma unroll
    for (int n2 = 0; n2 < 2; ++n2)
#pragma unroll
      for (int r = 0; r < 8; ++r)
        lp[(r + 8 * hl) * 32 + n2 * 16 + mrow] = (bf16_t)p[n2][r];

    asm volatile("s_wait_dscnt 0" ::: "memory");

    v16bf pf;
    {
      const bf16_t* pp = lp + mrow * 32 + kbaseA;
      v8bf a0 = *(const v8bf*)(pp);
      v8bf a1 = *(const v8bf*)(pp + 16);
      pf = cat8(a0, a1);
    }
    asm volatile("" ::: "memory");

    // ---- O += P V  (V pre-transposed [H][T] -> contiguous B-frag loads) ----
#pragma unroll
    for (int t = 0; t < 4; ++t) {
      const bf16_t* vp = vtb + (size_t)b * H_DIM * T_DIM
                       + (size_t)(t * 16 + mrow) * T_DIM + sbase + koffB;
      v8bf b0 = *(const v8bf*)(vp);
      v8bf b1 = *(const v8bf*)(vp + 16);
      v16bf vf = cat8(b0, b1);
      o[t] = __builtin_amdgcn_wmma_f32_16x16x32_bf16(
          false, pf, false, vf, (short)0, o[t], false, false);
    }
  }

  // ---- epilogue: normalize and write fp32 ----
  float* orow = out + ((size_t)b * T_DIM + qbase) * H_DIM;
#pragma unroll
  for (int r = 0; r < 8; ++r) {
    const float inv = 1.0f / lrun[r];
#pragma unroll
    for (int t = 0; t < 4; ++t)
      orow[(size_t)(r + 8 * hl) * H_DIM + t * 16 + mrow] = o[t][r] * inv;
  }
}

// ---------------- host launch ----------------
extern "C" void kernel_launch(void* const* d_in, const int* in_sizes, int n_in,
                              void* d_out, int out_size, void* d_ws, size_t ws_size,
                              hipStream_t stream) {
  const float* x  = (const float*)d_in[0];
  const float* Wq = (const float*)d_in[1];
  const float* bq = (const float*)d_in[2];
  const float* Wk = (const float*)d_in[3];
  const float* bk = (const float*)d_in[4];
  const float* Wv = (const float*)d_in[5];
  const float* bv = (const float*)d_in[6];
  float* out = (float*)d_out;

  bf16_t* wqt = (bf16_t*)d_ws;
  bf16_t* wkt = wqt + (size_t)C_DIM * H_DIM;
  bf16_t* wvt = wkt + (size_t)C_DIM * H_DIM;
  bf16_t* qb  = wvt + (size_t)C_DIM * H_DIM;
  bf16_t* kb  = qb  + (size_t)B_DIM * T_DIM * H_DIM;
  bf16_t* vtb = kb  + (size_t)B_DIM * T_DIM * H_DIM;

  const int wt_threads = C_DIM * H_DIM;                   // 65536
  wtrans_kernel<<<wt_threads / 256, 256, 0, stream>>>(Wq, wqt);
  wtrans_kernel<<<wt_threads / 256, 256, 0, stream>>>(Wk, wkt);
  wtrans_kernel<<<wt_threads / 256, 256, 0, stream>>>(Wv, wvt);

  const int n_row_tiles = (B_DIM * T_DIM) / 16;           // 1024 waves
  qkv_kernel<<<n_row_tiles / 8, 256, 0, stream>>>(x, wqt, wkt, wvt,
                                                  bq, bk, bv, qb, kb, vtb);

  const int n_attn_waves = B_DIM * (T_DIM / 16);          // 1024 waves
  attn_kernel<<<n_attn_waves / 8, 256, 0, stream>>>(qb, kb, vtb, out);
}
